// Block_ATC_46445776339632
// MI455X (gfx1250) — matrix operand order repaired and checked by reading, not compiled, not run on hardware
//
#include <hip/hip_runtime.h>
#include <math.h>

// ---------------------------------------------------------------------------
// ToMe ViT block for MI455X (gfx1250), wave32 + bf16 WMMA + TDM tile loads.
//   d_out = [ x:64*99*768 | size_new:64*99 | labels:64*197 ]  (floats)
// ---------------------------------------------------------------------------

typedef __attribute__((ext_vector_type(16))) __bf16 v16bf;
typedef __attribute__((ext_vector_type(8)))  float  v8f;
typedef unsigned int t_v4u __attribute__((ext_vector_type(4)));
typedef int          t_v4i __attribute__((ext_vector_type(4)));
typedef int          t_v8i __attribute__((ext_vector_type(8)));

#if defined(__HIP_DEVICE_COMPILE__)
#if defined(__has_builtin)
#if __has_builtin(__builtin_amdgcn_tensor_load_to_lds)
#define USE_TDM 1
#endif
#endif
#endif

union Frag16 {
  v16bf v;
  uint4 q[2];
  unsigned short u[16];
};

__device__ __forceinline__ unsigned short f2bf(float f) {
  unsigned int u = __float_as_uint(f);
  u += 0x7FFFu + ((u >> 16) & 1u);          // round-to-nearest-even
  return (unsigned short)(u >> 16);
}
__device__ __forceinline__ float bf2f(unsigned short h) {
  return __uint_as_float(((unsigned int)h) << 16);
}
__device__ __forceinline__ v8f wmma_bf16(v16bf a, v16bf b, v8f c) {
  return __builtin_amdgcn_wmma_f32_16x16x32_bf16(false, a, false, b,
                                                 (short)0, c, false, false);
}
__device__ __forceinline__ v8f v8f_zero() {
  v8f z = {0.f, 0.f, 0.f, 0.f, 0.f, 0.f, 0.f, 0.f};
  return z;
}

// ------------------------------- utility kernels ---------------------------

__global__ void cvt_f32_bf16_kernel(const float* __restrict__ in,
                                    unsigned short* __restrict__ out, int n) {
  int i = blockIdx.x * blockDim.x + threadIdx.x;
  for (; i < n; i += gridDim.x * blockDim.x) out[i] = f2bf(in[i]);
}

__global__ void zero_f32_kernel(float* __restrict__ p, int n) {
  int i = blockIdx.x * blockDim.x + threadIdx.x;
  for (; i < n; i += gridDim.x * blockDim.x) p[i] = 0.f;
}

__global__ void i32_to_f32_kernel(const int* __restrict__ in,
                                  float* __restrict__ out, int n) {
  int i = blockIdx.x * blockDim.x + threadIdx.x;
  for (; i < n; i += gridDim.x * blockDim.x) out[i] = (float)in[i];
}

// ------------------------------- LayerNorm ---------------------------------

__global__ __launch_bounds__(256)
void ln_bf16_kernel(const float* __restrict__ x, const float* __restrict__ g,
                    const float* __restrict__ bvec,
                    unsigned short* __restrict__ out, int C) {
  __shared__ float red[256];
  int row = blockIdx.x, tid = threadIdx.x;
  const float* xr = x + (size_t)row * C;

  float s = 0.f;
  for (int c = tid; c < C; c += 256) s += xr[c];
  red[tid] = s; __syncthreads();
  for (int o = 128; o > 0; o >>= 1) { if (tid < o) red[tid] += red[tid + o]; __syncthreads(); }
  float mean = red[0] / (float)C;
  __syncthreads();

  float v = 0.f;
  for (int c = tid; c < C; c += 256) { float d = xr[c] - mean; v += d * d; }
  red[tid] = v; __syncthreads();
  for (int o = 128; o > 0; o >>= 1) { if (tid < o) red[tid] += red[tid + o]; __syncthreads(); }
  float rstd = rsqrtf(red[0] / (float)C + 1e-5f);

  unsigned short* orow = out + (size_t)row * C;
  for (int c = tid; c < C; c += 256)
    orow[c] = f2bf((xr[c] - mean) * rstd * g[c] + bvec[c]);
}

// ------------------------------- WMMA GEMM ---------------------------------
// C[M,N] = A[M,K](bf16) * B[K,N](bf16)  (+bias)(+gelu)(+residual)
// 128x128 block tile, 8 waves, each wave 64x32 via 4x2 wmma_f32_16x16x32_bf16.
// A tile staged by the Tensor Data Mover (pad fields give the 40-elem LDS
// stride; tensor-dim OOB zero-fills M-edge rows). B tile staged+transposed by
// the VALU path (TDM cannot transpose).

#define GTM 128
#define GTN 128
#define GTK 32
#define LDP 40   // padded LDS row stride (elems): 32 data + 8 pad

__global__ __launch_bounds__(256)
void gemm_bf16_kernel(const unsigned short* __restrict__ A,
                      const unsigned short* __restrict__ B,
                      const float* __restrict__ bias,
                      const float* __restrict__ resid,
                      float* __restrict__ outF,
                      unsigned short* __restrict__ outH,
                      int M, int N, int K, int gelu) {
  __shared__ __align__(16) unsigned short As[GTM * LDP]; // [row][k], k-contig
  __shared__ __align__(16) unsigned short Bs[GTN * LDP]; // [col][k], k-contig
  int tid = threadIdx.x, lane = tid & 31, wid = tid >> 5;
  int bm = blockIdx.y * GTM, bn = blockIdx.x * GTN;
  int wm = (wid >> 2) * 64;   // 2 waves along M
  int wn = (wid & 3) * 32;    // 4 waves along N
  int m15 = lane & 15;
  int kbA = (lane < 16) ? 0 : 8;
  int kbB = (lane < 16) ? 0 : 16;

  v8f acc[4][2];
  for (int i = 0; i < 4; ++i) for (int j = 0; j < 2; ++j) acc[i][j] = v8f_zero();

  for (int k0 = 0; k0 < K; k0 += GTK) {
    __syncthreads();  // LDS consumed by previous iteration's fragments

#ifdef USE_TDM
    if (wid == 0) {
      // ---- TDM descriptor: 2D tile [128 rows x 32 elems] of bf16 ----
      unsigned long long ga =
          (unsigned long long)(const void*)(A + (size_t)bm * K + k0);
      unsigned int lds = (unsigned int)(unsigned long long)(const void*)As;
      unsigned int td0 = (unsigned int)(K - k0);            // dim0 extent left
      unsigned int td1 = (unsigned int)(M - bm);            // rows left (OOB->0)
      unsigned long long st0 = (unsigned long long)K;       // dim0 stride (elems)
      // group0: count=1 | lds_addr | global_addr[56:0] | type=2
      t_v4u g0 = { 1u, lds, (unsigned int)ga,
                   (unsigned int)((ga >> 32) & 0x01FFFFFFu) | (2u << 30) };
      // group1: data_size=1(2B), pad_enable, pad_interval=3(16 dw),
      //         pad_amount=3(4 dw); dims/strides packed per D# bit layout.
      t_v8i g1 = {
        (int)((1u << 16) | (1u << 20) | (3u << 22) | (3u << 25)),
        (int)((td0 & 0xFFFFu) << 16),
        (int)((td0 >> 16) | ((td1 & 0xFFFFu) << 16)),
        (int)((td1 >> 16) | (32u << 16)),                    // tile_dim0 = 32
        (int)128,                                            // tile_dim1 = 128
        (int)(st0 & 0xFFFFFFFFu),
        (int)((st0 >> 32) & 0xFFFFu),
        0 };
      t_v4i gz = { 0, 0, 0, 0 };
#if __has_include(<hip/amd_detail/amd_gfx1250_TDM.h>)
      t_v8i gz8 = { 0, 0, 0, 0, 0, 0, 0, 0 };
      __builtin_amdgcn_tensor_load_to_lds(g0, g1, gz, gz, gz8, 0);
#else
      __builtin_amdgcn_tensor_load_to_lds(g0, g1, gz, gz, 0);
#endif
    }
#else
    { // fallback: cooperative A tile 128x32, 16 bf16 per thread
      int r = tid >> 1, cs = (tid & 1) * 16;
      int gr = bm + r; if (gr >= M) gr = M - 1;
      const uint4* src = (const uint4*)(A + (size_t)gr * K + k0 + cs);
      uint4 u0 = src[0], u1 = src[1];
      *(uint4*)(As + r * LDP + cs)     = u0;
      *(uint4*)(As + r * LDP + cs + 8) = u1;
    }
#endif

    { // B tile: 32x128 read row-major, stored transposed [n][k]
      int kk = tid >> 3, ns = (tid & 7) * 16;
      const unsigned short* src = B + (size_t)(k0 + kk) * N + bn + ns;
      Frag16 t; t.q[0] = *(const uint4*)src; t.q[1] = *(const uint4*)(src + 8);
      for (int e = 0; e < 16; ++e) Bs[(ns + e) * LDP + kk] = t.u[e];
    }
    if (k0 + GTK < K)  // prefetch next B slab (global_prefetch_b8)
      __builtin_prefetch(B + (size_t)(k0 + GTK + (tid >> 3)) * N + bn, 0, 1);

#ifdef USE_TDM
    if (wid == 0) __builtin_amdgcn_s_wait_tensorcnt(0);
#endif
    __syncthreads();

    Frag16 a[4], b[2];
    for (int s = 0; s < 4; ++s) {  // A frag: two contiguous 8-elem runs
      const unsigned short* p = As + (wm + s * 16 + m15) * LDP;
      a[s].q[0] = *(const uint4*)(p + kbA);
      a[s].q[1] = *(const uint4*)(p + kbA + 16);
    }
    for (int s = 0; s < 2; ++s) {  // B frag: two contiguous 8-elem runs
      const unsigned short* p = Bs + (wn + s * 16 + m15) * LDP + kbA;
      b[s].q[0] = *(const uint4*)p;
      b[s].q[1] = *(const uint4*)(p + 16);
    }
    for (int i = 0; i < 4; ++i)
      for (int j = 0; j < 2; ++j)
        acc[i][j] = wmma_bf16(a[i].v, b[j].v, acc[i][j]);
  }

  // Epilogue: C layout lane<16 -> M=r, lane>=16 -> M=r+8; N = lane&15
  for (int i = 0; i < 4; ++i)
    for (int j = 0; j < 2; ++j)
      for (int r = 0; r < 8; ++r) {
        int row = bm + wm + i * 16 + r + ((lane < 16) ? 0 : 8);
        int col = bn + wn + j * 16 + m15;
        if (row < M) {
          float v = acc[i][j][r];
          if (bias)  v += bias[col];
          if (gelu)  v = 0.5f * v * (1.f + erff(v * 0.70710678118f));
          if (resid) v += resid[(size_t)row * N + col];
          if (outF) outF[(size_t)row * N + col] = v;
          else      outH[(size_t)row * N + col] = f2bf(v);
        }
      }
}

// ------------------------------- Attention ---------------------------------
// One block per (32-query tile, head, batch). S and Vt tiles in LDS (~60KB).
// qkv layout: row=b*197+n, cols [q:0..767 | k:768..1535 | v:1536..2303].

#define ATN 197
#define ATP 224   // padded key count (14 tiles of 16; multiple of 32)

__global__ __launch_bounds__(256)
void attn_kernel(const unsigned short* __restrict__ qkv,
                 const float* __restrict__ asize,
                 unsigned short* __restrict__ xa) {
  __shared__ __align__(16) unsigned short Qs[32 * 64];
  __shared__ float Sc[32 * ATP];
  __shared__ __align__(16) unsigned short Vt[64 * ATP]; // [d][key]

  int tid = threadIdx.x, lane = tid & 31, wid = tid >> 5;
  int qb = blockIdx.x * 32, h = blockIdx.y, b = blockIdx.z;
  int m15 = lane & 15;
  int kbA = (lane < 16) ? 0 : 8;
  int kbB = (lane < 16) ? 0 : 16;
  const size_t rowB = (size_t)b * ATN;

  { // Q tile 32x64 bf16
    int r = tid >> 3, cs = (tid & 7) * 8;
    int q = qb + r;
    if (q < ATN) {
      const uint4* s = (const uint4*)(qkv + (rowB + q) * 2304 + h * 64 + cs);
      *(uint4*)(Qs + r * 64 + cs) = *s;
    } else {
      uint4 z{0, 0, 0, 0};
      *(uint4*)(Qs + r * 64 + cs) = z;
    }
  }
  if (tid < ATP) { // V tile transposed: Vt[d][key]
    int key = tid;
    if (key < ATN) {
      const unsigned short* s = qkv + (rowB + key) * 2304 + 1536 + h * 64;
      for (int d = 0; d < 64; ++d) Vt[d * ATP + key] = s[d];
    } else {
      for (int d = 0; d < 64; ++d) Vt[d * ATP + key] = 0;
    }
  }
  __syncthreads();

  // --- scores S = Q K^T * 0.125 + log(size), via WMMA ---
  int wq = wid >> 2;                      // query subtile 0..1
  for (int jt = (wid & 3); jt < 14; jt += 4) {
    v8f acc = v8f_zero();
    int key = jt * 16 + m15;
    int keyc = (key < ATN) ? key : (ATN - 1);
    for (int ch = 0; ch < 2; ++ch) {      // d = ch*32 .. +31
      Frag16 a, bk;
      const unsigned short* ap = Qs + (wq * 16 + m15) * 64 + ch * 32 + kbA;
      a.q[0] = *(const uint4*)ap;
      a.q[1] = *(const uint4*)(ap + 16);
      const unsigned short* bp =
          qkv + (rowB + keyc) * 2304 + 768 + h * 64 + ch * 32 + kbA;
      bk.q[0] = *(const uint4*)bp;
      bk.q[1] = *(const uint4*)(bp + 16);
      acc = wmma_bf16(a.v, bk.v, acc);
    }
    float lsz = (key < ATN) ? logf(asize[rowB + key]) : 0.f;
    for (int r = 0; r < 8; ++r) {
      int row = wq * 16 + r + ((lane < 16) ? 0 : 8);
      Sc[row * ATP + key] = (key < ATN) ? (acc[r] * 0.125f + lsz) : -1e30f;
    }
  }
  __syncthreads();

  // --- softmax (one thread per query row; rows are short) ---
  if (tid < 32) {
    float* row = Sc + tid * ATP;
    float mx = -1e30f;
    for (int c = 0; c < ATN; ++c) mx = fmaxf(mx, row[c]);
    float s = 0.f;
    for (int c = 0; c < ATN; ++c) { float e = __expf(row[c] - mx); row[c] = e; s += e; }
    float inv = 1.f / s;
    for (int c = 0; c < ATN; ++c) row[c] *= inv;
    for (int c = ATN; c < ATP; ++c) row[c] = 0.f;
  }
  __syncthreads();

  // --- out = P V via WMMA (reduce over 224 padded keys, 7 k-steps) ---
  int dt = wid & 3;                       // d subtile 0..3
  v8f acc = v8f_zero();
  for (int st = 0; st < 7; ++st) {
    Frag16 a, bv;
    int arow = wq * 16 + m15;
    for (int i = 0; i < 16; ++i) {
      int key = st * 32 + kbA + (i & 7) + ((i >> 3) << 4);
      a.u[i] = f2bf(Sc[arow * ATP + key]);
    }
    const unsigned short* bp = Vt + (dt * 16 + m15) * ATP + st * 32 + kbA;
    bv.q[0] = *(const uint4*)bp;
    bv.q[1] = *(const uint4*)(bp + 16);
    acc = wmma_bf16(a.v, bv.v, acc);
  }
  for (int r = 0; r < 8; ++r) {
    int q = qb + wq * 16 + r + ((lane < 16) ? 0 : 8);
    if (q < ATN)
      xa[(rowB + q) * 768 + h * 64 + dt * 16 + m15] = f2bf(acc[r]);
  }
}

// ----------------------------- metric / dist -------------------------------

__global__ __launch_bounds__(64)
void metric_kernel(const unsigned short* __restrict__ qkv,
                   float* __restrict__ mn) {
  __shared__ float sm[64];
  int row = blockIdx.x;          // b*197+n
  int d = threadIdx.x;
  float s = 0.f;
  for (int h = 0; h < 12; ++h)
    s += bf2f(qkv[(size_t)row * 2304 + 768 + h * 64 + d]);
  s *= (1.f / 12.f);
  sm[d] = s * s; __syncthreads();
  for (int o = 32; o > 0; o >>= 1) { if (d < o) sm[d] += sm[d + o]; __syncthreads(); }
  mn[(size_t)row * 64 + d] = s * rsqrtf(sm[0]);
}

__global__ __launch_bounds__(256)
void dist_kernel(const float* __restrict__ mn, float* __restrict__ dist) {
  __shared__ float Ar[16][64];
  int b = blockIdx.y, i0 = blockIdx.x * 16, tid = threadIdx.x;
  for (int e = tid; e < 16 * 64; e += 256) {
    int i = e >> 6, d = e & 63, ii = i0 + i;
    Ar[i][d] = (ii < 196) ? mn[((size_t)b * 197 + 1 + ii) * 64 + d] : 0.f;
  }
  __syncthreads();
  for (int j = tid; j < 196; j += 256) {
    const float* bp = mn + ((size_t)b * 197 + 1 + j) * 64;
    float acc[16];
    for (int i = 0; i < 16; ++i) acc[i] = 0.f;
    for (int d = 0; d < 64; ++d) {
      float bv = bp[d];
      for (int i = 0; i < 16; ++i) acc[i] += Ar[i][d] * bv;
    }
    for (int i = 0; i < 16; ++i) {
      int ii = i0 + i;
      if (ii < 196) dist[((size_t)b * 196 + ii) * 196 + j] = 1.f - acc[i];
    }
  }
}

// --------------------------- agglomerative clustering ----------------------
// One workgroup per batch; the full 196x196 distance matrix + cached per-row
// minima live in dynamic LDS (~161 KB, exploiting CDNA5's 320 KB WGP LDS).

#define AGG_T   196
#define AGG_INF 1e10f

__global__ __launch_bounds__(256)
void agglom_kernel(const float* __restrict__ dist, int* __restrict__ lout) {
  extern __shared__ float smraw[];
  float* D      = smraw;                       // 196*196
  float* rowMin = D + AGG_T * AGG_T;           // 196
  float* sizes  = rowMin + AGG_T;              // 196
  float* redV   = sizes + AGG_T;               // 256
  int*   rowArg = (int*)(redV + 256);          // 196
  int*   labels = rowArg + AGG_T;              // 196
  int*   act    = labels + AGG_T;              // 196
  int*   redI   = act + AGG_T;                 // 256
  int*   redJ   = redI + 256;                  // 256
  __shared__ int s_i, s_j;
  __shared__ float s_ni, s_nj;
  __shared__ int rank[AGG_T];

  int b = blockIdx.x, tid = threadIdx.x;
  const float* src = dist + (size_t)b * AGG_T * AGG_T;
  for (int e = tid; e < AGG_T * AGG_T; e += 256) {
    int i = e / AGG_T, j = e - i * AGG_T;
    D[e] = (i == j) ? AGG_INF : src[e];
  }
  for (int t = tid; t < AGG_T; t += 256) { sizes[t] = 1.f; labels[t] = t; act[t] = 1; }
  __syncthreads();
  for (int t = tid; t < AGG_T; t += 256) {
    float mv = 1e30f; int ma = 0;
    const float* r = D + t * AGG_T;
    for (int c = 0; c < AGG_T; ++c)
      if (c != t && r[c] < mv) { mv = r[c]; ma = c; }
    rowMin[t] = mv; rowArg[t] = ma;
  }
  __syncthreads();

  for (int it = 0; it < AGG_T - 98; ++it) {   // 98 merges
    // global argmin over cached row minima
    float mv = 1e30f; int mi = -1, mj = -1;
    for (int t = tid; t < AGG_T; t += 256)
      if (act[t] && rowMin[t] < mv) { mv = rowMin[t]; mi = t; mj = rowArg[t]; }
    redV[tid] = mv; redI[tid] = mi; redJ[tid] = mj; __syncthreads();
    for (int o = 128; o > 0; o >>= 1) {
      if (tid < o) {
        bool take = (redV[tid + o] < redV[tid]) ||
                    (redV[tid + o] == redV[tid] && redI[tid + o] >= 0 &&
                     (redI[tid] < 0 || redI[tid + o] < redI[tid]));
        if (take) { redV[tid] = redV[tid + o]; redI[tid] = redI[tid + o]; redJ[tid] = redJ[tid + o]; }
      }
      __syncthreads();
    }
    if (tid == 0) {
      int a = redI[0], c = redJ[0];
      s_i = (a < c) ? a : c; s_j = (a < c) ? c : a;
      s_ni = sizes[s_i]; s_nj = sizes[s_j];
    }
    __syncthreads();
    int i = s_i, j = s_j;
    float ni = s_ni, nj = s_nj, inv = 1.f / (ni + nj);

    // Lance-Williams average-linkage row update
    for (int t = tid; t < AGG_T; t += 256) {
      if (act[t] && t != i && t != j) {
        float nd = (ni * D[i * AGG_T + t] + nj * D[j * AGG_T + t]) * inv;
        D[i * AGG_T + t] = nd; D[t * AGG_T + i] = nd;
        D[j * AGG_T + t] = AGG_INF; D[t * AGG_T + j] = AGG_INF;
      }
      if (labels[t] == j) labels[t] = i;
    }
    __syncthreads();
    if (tid == 0) {
      D[i * AGG_T + i] = AGG_INF; D[i * AGG_T + j] = AGG_INF; D[j * AGG_T + i] = AGG_INF;
      sizes[i] = ni + nj; act[j] = 0;
    }
    __syncthreads();

    // incremental row-min maintenance (rescan only when invalidated)
    for (int t = tid; t < AGG_T; t += 256) {
      if (!act[t]) continue;
      if (t == i) {
        float bv = 1e30f; int ba = 0;
        const float* r = D + i * AGG_T;
        for (int c = 0; c < AGG_T; ++c)
          if (act[c] && c != i && r[c] < bv) { bv = r[c]; ba = c; }
        rowMin[t] = bv; rowArg[t] = ba;
      } else {
        float nd = D[t * AGG_T + i];
        if (nd < rowMin[t]) { rowMin[t] = nd; rowArg[t] = i; }
        else if (rowArg[t] == i || rowArg[t] == j) {
          float bv = 1e30f; int ba = 0;
          const float* r = D + t * AGG_T;
          for (int c = 0; c < AGG_T; ++c)
            if (act[c] && c != t && r[c] < bv) { bv = r[c]; ba = c; }
          rowMin[t] = bv; rowArg[t] = ba;
        }
      }
    }
    __syncthreads();
  }

  if (tid == 0) { int c = 0; for (int t = 0; t < AGG_T; ++t) rank[t] = act[t] ? c++ : -1; }
  __syncthreads();
  if (tid == 0) lout[b * 197] = 0;
  for (int t = tid; t < AGG_T; t += 256)
    lout[b * 197 + 1 + t] = rank[labels[t]] + 1;
}

// ------------------------------- token merge -------------------------------

__global__ __launch_bounds__(256)
void merge_scatter_kernel(const float* __restrict__ x1,
                          const float* __restrict__ asize,
                          const int* __restrict__ labels,
                          float* __restrict__ xs, float* __restrict__ sn) {
  int b = blockIdx.y, n = blockIdx.x;
  int lbl = labels[b * 197 + n];
  float sz = asize[b * 197 + n];
  const float* src = x1 + ((size_t)b * 197 + n) * 768;
  float* dst = xs + ((size_t)b * 99 + lbl) * 768;
  for (int c = threadIdx.x; c < 768; c += 256) atomicAdd(&dst[c], src[c] * sz);
  if (threadIdx.x == 0) atomicAdd(&sn[b * 99 + lbl], sz);
}

__global__ __launch_bounds__(256)
void merge_div_kernel(const float* __restrict__ xs, const float* __restrict__ sn,
                      float* __restrict__ x2, float* __restrict__ out_sz) {
  int b = blockIdx.y, k = blockIdx.x;
  float s = sn[b * 99 + k];
  float inv = 1.f / s;
  const float* src = xs + ((size_t)b * 99 + k) * 768;
  float* dst = x2 + ((size_t)b * 99 + k) * 768;
  for (int c = threadIdx.x; c < 768; c += 256) dst[c] = src[c] * inv;
  if (threadIdx.x == 0) out_sz[b * 99 + k] = s;
}

// ------------------------------- launch ------------------------------------

extern "C" void kernel_launch(void* const* d_in, const int* in_sizes, int n_in,
                              void* d_out, int out_size, void* d_ws, size_t ws_size,
                              hipStream_t stream) {
  (void)in_sizes; (void)n_in; (void)out_size; (void)ws_size;
  const float* x     = (const float*)d_in[0];
  const float* asize = (const float*)d_in[1];
  const float* n1g   = (const float*)d_in[2];
  const float* n1b   = (const float*)d_in[3];
  const float* wqkv  = (const float*)d_in[4];
  const float* wproj = (const float*)d_in[5];
  const float* bproj = (const float*)d_in[6];
  const float* n2g   = (const float*)d_in[7];
  const float* n2b   = (const float*)d_in[8];
  const float* wfc1  = (const float*)d_in[9];
  const float* bfc1  = (const float*)d_in[10];
  const float* wfc2  = (const float*)d_in[11];
  const float* bfc2  = (const float*)d_in[12];

  const int Bb = 64, Nn = 197, Cc = 768, C3 = 2304, Hf = 3072, Kt = 99;
  const int M1 = Bb * Nn;   // 12608
  const int M2 = Bb * Kt;   // 6336

  char* ws = (char*)d_ws;
  size_t off = 0;
  auto take = [&](size_t bytes) {
    char* p = ws + off;
    off += (bytes + 255) & ~(size_t)255;
    return p;
  };
  unsigned short* wqkv_h  = (unsigned short*)take((size_t)Cc * C3 * 2);
  unsigned short* wproj_h = (unsigned short*)take((size_t)Cc * Cc * 2);
  unsigned short* wfc1_h  = (unsigned short*)take((size_t)Cc * Hf * 2);
  unsigned short* wfc2_h  = (unsigned short*)take((size_t)Hf * Cc * 2);
  unsigned short* h1_h    = (unsigned short*)take((size_t)M1 * Cc * 2);
  unsigned short* qkv_h   = (unsigned short*)take((size_t)M1 * C3 * 2);
  unsigned short* xa_h    = (unsigned short*)take((size_t)M1 * Cc * 2);
  float*          x1_f    = (float*)take((size_t)M1 * Cc * 4);
  float*          mn_f    = (float*)take((size_t)M1 * 64 * 4);
  float*          dist_f  = (float*)take((size_t)Bb * 196 * 196 * 4);
  int*            lbl_i   = (int*)take((size_t)Bb * Nn * 4);
  float*          xs_f    = (float*)take((size_t)Bb * Kt * Cc * 4);
  float*          sn_f    = (float*)take((size_t)Bb * Kt * 4);
  float*          x2_f    = (float*)take((size_t)Bb * Kt * Cc * 4);
  unsigned short* h2_h    = (unsigned short*)take((size_t)M2 * Cc * 2);
  unsigned short* mid_h   = (unsigned short*)take((size_t)M2 * Hf * 2);

  float* out_x  = (float*)d_out;
  float* out_sz = out_x + (size_t)Bb * Kt * Cc;   // 4,866,048
  float* out_lb = out_sz + (size_t)Bb * Kt;       // +6,336

  // 0) weights -> bf16
  cvt_f32_bf16_kernel<<<512, 256, 0, stream>>>(wqkv, wqkv_h, Cc * C3);
  cvt_f32_bf16_kernel<<<512, 256, 0, stream>>>(wproj, wproj_h, Cc * Cc);
  cvt_f32_bf16_kernel<<<512, 256, 0, stream>>>(wfc1, wfc1_h, Cc * Hf);
  cvt_f32_bf16_kernel<<<512, 256, 0, stream>>>(wfc2, wfc2_h, Hf * Cc);

  // 1) LN1 -> bf16
  ln_bf16_kernel<<<M1, 256, 0, stream>>>(x, n1g, n1b, h1_h, Cc);

  // 2) QKV GEMM (bf16 out)
  gemm_bf16_kernel<<<dim3(C3 / 128, (M1 + 127) / 128), 256, 0, stream>>>(
      h1_h, wqkv_h, nullptr, nullptr, nullptr, qkv_h, M1, C3, Cc, 0);

  // 3) fused attention
  attn_kernel<<<dim3(7, 12, Bb), 256, 0, stream>>>(qkv_h, asize, xa_h);

  // 4) proj GEMM + bias + residual -> x1 (fp32)
  gemm_bf16_kernel<<<dim3(Cc / 128, (M1 + 127) / 128), 256, 0, stream>>>(
      xa_h, wproj_h, bproj, x, x1_f, nullptr, M1, Cc, Cc, 0);

  // 5) metric (head-mean of K, L2-normalized) and cosine distances
  metric_kernel<<<M1, 64, 0, stream>>>(qkv_h, mn_f);
  dist_kernel<<<dim3(13, Bb), 256, 0, stream>>>(mn_f, dist_f);

  // 6) agglomerative clustering in big LDS
  size_t agg_smem = ((size_t)196 * 196 + 196 * 2 + 256) * 4 + (196 * 3 + 512) * 4;
  agglom_kernel<<<Bb, 256, agg_smem, stream>>>(dist_f, lbl_i);

  // 7) weighted-average token merge
  zero_f32_kernel<<<1024, 256, 0, stream>>>(xs_f, Bb * Kt * Cc);
  zero_f32_kernel<<<32, 256, 0, stream>>>(sn_f, Bb * Kt);
  merge_scatter_kernel<<<dim3(Nn, Bb), 256, 0, stream>>>(x1_f, asize, lbl_i, xs_f, sn_f);
  merge_div_kernel<<<dim3(Kt, Bb), 256, 0, stream>>>(xs_f, sn_f, x2_f, out_sz);

  // 8) MLP: LN2 -> fc1(+gelu, bf16) -> fc2(+bias+residual, fp32 -> d_out)
  ln_bf16_kernel<<<M2, 256, 0, stream>>>(x2_f, n2g, n2b, h2_h, Cc);
  gemm_bf16_kernel<<<dim3(Hf / 128, (M2 + 127) / 128), 256, 0, stream>>>(
      h2_h, wfc1_h, bfc1, nullptr, nullptr, mid_h, M2, Hf, Cc, 1);
  gemm_bf16_kernel<<<dim3(Cc / 128, (M2 + 127) / 128), 256, 0, stream>>>(
      mid_h, wfc2_h, bfc2, x2_f, out_x, nullptr, M2, Cc, Hf, 0);

  // 9) labels (int) -> float tail of d_out
  i32_to_f32_kernel<<<50, 256, 0, stream>>>(lbl_i, out_lb, Bb * Nn);
}
